// OutputModule_82738249990287
// MI455X (gfx1250) — matrix operand order repaired and checked by reading, not compiled, not run on hardware
//
#include <hip/hip_runtime.h>
#include <hip/hip_bf16.h>
#include <math.h>

typedef __bf16 bf16;
typedef __attribute__((ext_vector_type(16))) __bf16 v16bf;
typedef __attribute__((ext_vector_type(8)))  float  v8f;

// Problem constants (fixed by the reference)
constexpr int NN = 128;
constexpr int BB = 2;
constexpr int EE = 512;
constexpr int HH = 1024;
constexpr int GG = 50;
constexpr int LL = 3;
constexpr int KF = 3 * EE;          // 1536 input features
constexpr int NPAIR = NN * NN * BB; // 32768
constexpr int MT = 32;              // pairs per workgroup (two 16-row WMMA M tiles)
constexpr float RBF_RADIUS = 12.0f;
constexpr float INV_AVG    = 1.0f / 60.0f;
constexpr float INV_AVG2   = 1.0f / 3600.0f;

// ---------------- prep kernels: fp32 -> bf16 (transposed) weights ----------------

__global__ void transpose_to_bf16(const float* __restrict__ in, bf16* __restrict__ out,
                                  int K, int Ncol) {
  // in: K x Ncol (row major), out: Ncol x K (row major)  => out[n][k] = in[k][n]
  int idx = blockIdx.x * blockDim.x + threadIdx.x;
  if (idx < K * Ncol) {
    int k = idx / Ncol, n = idx % Ncol;
    out[(size_t)n * K + k] = (bf16)in[idx];
  }
}

__global__ void convert_to_bf16(const float* __restrict__ in, bf16* __restrict__ out, int n) {
  int i = blockIdx.x * blockDim.x + threadIdx.x;
  if (i < n) out[i] = (bf16)in[i];
}

__global__ void zero_f32(float* __restrict__ out, int n) {
  int i = blockIdx.x * blockDim.x + threadIdx.x;
  if (i < n) out[i] = 0.0f;
}

// ---------------- fused all-pairs MLP kernel ----------------

__device__ __forceinline__ v16bf load_a_frag(const bf16* row, int ks, int lane) {
  // 16-bit A layout: lane holds M=lane&15; elems 0..7 = K+0..7, 8..15 = K+16..23,
  // with K base offset by 8 for lanes 16..31.
  const bf16* p = row + ks * 32 + ((lane >> 4) << 3);
  v16bf a;
  *(uint4*)&a       = *(const uint4*)p;        // K+0..7
  *((uint4*)&a + 1) = *(const uint4*)(p + 16); // K+16..23
  return a;
}

__device__ __forceinline__ v16bf load_b_frag(const bf16* wt_row, int ks, int lane) {
  // 16-bit B layout: lane holds column n=lane&15; elems 0..7 = K+0..7, 8..15 = K+16..23,
  // with K base offset by 8 for lanes 16..31. wt_row points at W^T[n].
  const bf16* p = wt_row + ks * 32 + ((lane >> 4) << 3);
  v16bf b;
  *(uint4*)&b       = *(const uint4*)p;        // K+0..7
  *((uint4*)&b + 1) = *(const uint4*)(p + 16); // K+16..23
  return b;
}

__device__ __forceinline__ float gelu_exact(float v) {
  return 0.5f * v * (1.0f + erff(v * 0.70710678118654752f));
}

__global__ __launch_bounds__(256, 1)
void pair_mlp_kernel(const float* __restrict__ x,          // (N,B,E)
                     const float* __restrict__ dist,       // (N,N,B)
                     const float* __restrict__ vec_hat,    // (N,N,B,3)
                     const unsigned char* __restrict__ mask, // (B,N) bool
                     const float* __restrict__ rbf_W,      // (G,E)
                     const float* __restrict__ rbf_b,      // (E)
                     const bf16* __restrict__ eWinT, const float* __restrict__ e_bin,
                     const bf16* __restrict__ eWhT,  const float* __restrict__ e_bh,
                     const bf16* __restrict__ eWout, const float* __restrict__ e_bout,
                     const bf16* __restrict__ fWinT, const float* __restrict__ f_bin,
                     const bf16* __restrict__ fWhT,  const float* __restrict__ f_bh,
                     const bf16* __restrict__ fWout, const float* __restrict__ f_bout,
                     float* __restrict__ out)              // [0..1]=energy, [2..]=forces
{
  extern __shared__ __align__(16) char smem[];
  bf16* feat  = (bf16*)smem;                                 // MT x KF   (96 KB)
  bf16* hbuf  = (bf16*)(smem + (size_t)MT * KF * 2);         // MT x HH   (64 KB)
  float* esum = (float*)(smem + (size_t)MT * KF * 2 + (size_t)MT * HH * 2);

  const int tid  = threadIdx.x;
  const int wave = tid >> 5;    // 0..7
  const int lane = tid & 31;
  const int p0   = blockIdx.x * MT;

  if (tid < BB) esum[tid] = 0.0f;

  // ---------- stage 1: build feature tile [x_i | x_j | rbf@rbf_W] in bf16 ----------
  // 256 threads, 32 pairs -> two halves, 16 threads per pair
  for (int half = 0; half < 2; ++half) {
    const int pr  = half * 16 + (tid >> 4);  // pair row 0..31
    const int sub = tid & 15;                // 16 threads per pair
    const int p   = p0 + pr;
    const int b   = p & (BB - 1);
    const int ij  = p / BB;
    const int j   = ij & (NN - 1);
    const int i   = ij >> 7;   // N == 128

    const float* xi = x + ((size_t)i * BB + b) * EE;
    const float* xj = x + ((size_t)j * BB + b) * EE;
    bf16* frow = feat + (size_t)pr * KF;
    #pragma unroll 8
    for (int t = 0; t < 32; ++t) {
      int e = sub * 32 + t;
      frow[e]      = (bf16)xi[e];
      frow[EE + e] = (bf16)xj[e];
    }
    // RBF expansion (G=50 gaussians) projected to E, slice of 32 columns per thread
    const float d = dist[(size_t)(i * NN + j) * BB + b];
    const float delta = RBF_RADIUS / (float)(GG - 1);
    const float coeff = -0.5f / (delta * delta);
    float acc[32];
    #pragma unroll 8
    for (int t = 0; t < 32; ++t) acc[t] = rbf_b[sub * 32 + t];
    for (int k = 0; k < GG; ++k) {
      float dv = d - (float)k * delta;
      float gv = expf(coeff * dv * dv);
      const float* wr = rbf_W + (size_t)k * EE + sub * 32;
      #pragma unroll 8
      for (int t = 0; t < 32; ++t) acc[t] = fmaf(gv, wr[t], acc[t]);
    }
    #pragma unroll 8
    for (int t = 0; t < 32; ++t) frow[2 * EE + sub * 32 + t] = (bf16)acc[t];
  }
  __syncthreads();

  // ---------- stage 2: run both ResMLPs over the 32-pair tile ----------
  // Each wave owns a 128-column slice for BOTH 16-row M tiles: every B fragment
  // fetched from L2 feeds two WMMAs (halves weight traffic vs one-tile waves).
  const int nb   = wave * 128;        // 128-column slice, 8 WMMA N tiles
  const int col  = lane & 15;
  const int rowb = (lane >> 4) << 3;  // 0 or 8 (row base within a 16-row tile)

  for (int mlp = 0; mlp < 2; ++mlp) {
    const bf16*  WinT = mlp ? fWinT : eWinT;
    const float* bin  = mlp ? f_bin : e_bin;
    const bf16*  WhT  = mlp ? fWhT  : eWhT;
    const float* bh   = mlp ? f_bh  : e_bh;
    const bf16*  Wout = mlp ? fWout : eWout;
    const float  bout = mlp ? f_bout[0] : e_bout[0];

    // ----- layer 0: h = gelu(feat @ Win + bin), K = 1536 -----
    {
      v8f acc0[8], acc1[8];
      #pragma unroll
      for (int t = 0; t < 8; ++t) {
        float bv = bin[nb + t * 16 + col];
        #pragma unroll
        for (int r = 0; r < 8; ++r) { acc0[t][r] = bv; acc1[t][r] = bv; }
      }
      const bf16* arow0 = feat + (size_t)col * KF;          // M tile 0 (rows 0..15)
      const bf16* arow1 = feat + (size_t)(16 + col) * KF;   // M tile 1 (rows 16..31)
      for (int ks = 0; ks < KF / 32; ++ks) {
        v16bf a0 = load_a_frag(arow0, ks, lane);
        v16bf a1 = load_a_frag(arow1, ks, lane);
        #pragma unroll
        for (int t = 0; t < 8; ++t) {
          const bf16* wrow = WinT + (size_t)(nb + t * 16 + col) * KF;
          v16bf b = load_b_frag(wrow, ks, lane);
          acc0[t] = __builtin_amdgcn_wmma_f32_16x16x32_bf16(
                      false, a0, false, b, (short)0, acc0[t], false, false);
          acc1[t] = __builtin_amdgcn_wmma_f32_16x16x32_bf16(
                      false, a1, false, b, (short)0, acc1[t], false, false);
        }
      }
      #pragma unroll
      for (int t = 0; t < 8; ++t) {
        int n = nb + t * 16 + col;
        #pragma unroll
        for (int r = 0; r < 8; ++r) {
          hbuf[(size_t)(rowb + r) * HH + n]      = (bf16)gelu_exact(acc0[t][r]);
          hbuf[(size_t)(16 + rowb + r) * HH + n] = (bf16)gelu_exact(acc1[t][r]);
        }
      }
    }
    __syncthreads();

    // ----- layers 1..3: h = h + gelu(h @ Wh[l] + bh[l]), K = 1024 -----
    for (int l = 0; l < LL; ++l) {
      const bf16*  W  = WhT + (size_t)l * HH * HH;
      const float* bl = bh + l * HH;
      v8f acc0[8], acc1[8];
      #pragma unroll
      for (int t = 0; t < 8; ++t) {
        float bv = bl[nb + t * 16 + col];
        #pragma unroll
        for (int r = 0; r < 8; ++r) { acc0[t][r] = bv; acc1[t][r] = bv; }
      }
      const bf16* arow0 = hbuf + (size_t)col * HH;
      const bf16* arow1 = hbuf + (size_t)(16 + col) * HH;
      for (int ks = 0; ks < HH / 32; ++ks) {
        v16bf a0 = load_a_frag(arow0, ks, lane);
        v16bf a1 = load_a_frag(arow1, ks, lane);
        #pragma unroll
        for (int t = 0; t < 8; ++t) {
          const bf16* wrow = W + (size_t)(nb + t * 16 + col) * HH;
          v16bf b = load_b_frag(wrow, ks, lane);
          acc0[t] = __builtin_amdgcn_wmma_f32_16x16x32_bf16(
                      false, a0, false, b, (short)0, acc0[t], false, false);
          acc1[t] = __builtin_amdgcn_wmma_f32_16x16x32_bf16(
                      false, a1, false, b, (short)0, acc1[t], false, false);
        }
      }
      __syncthreads();  // all waves done reading hbuf before residual overwrite
      #pragma unroll
      for (int t = 0; t < 8; ++t) {
        int n = nb + t * 16 + col;
        #pragma unroll
        for (int r = 0; r < 8; ++r) {
          size_t i0 = (size_t)(rowb + r) * HH + n;
          size_t i1 = (size_t)(16 + rowb + r) * HH + n;
          hbuf[i0] = (bf16)((float)hbuf[i0] + gelu_exact(acc0[t][r]));
          hbuf[i1] = (bf16)((float)hbuf[i1] + gelu_exact(acc1[t][r]));
        }
      }
      __syncthreads();
    }

    // ----- head: scalar = h . Wout + bout, then scatter/reduce outputs -----
    {
      const int pr  = tid >> 3;   // 0..31
      const int sub = tid & 7;    // 8 threads per pair, 128 elements each
      const bf16* hrow = hbuf + (size_t)pr * HH + sub * 128;
      const bf16* wv   = Wout + sub * 128;
      float s = 0.0f;
      #pragma unroll 16
      for (int e = 0; e < 128; ++e) s = fmaf((float)hrow[e], (float)wv[e], s);
      #pragma unroll
      for (int off = 4; off > 0; off >>= 1) s += __shfl_xor(s, off, 8);

      if (sub == 0) {
        const int p  = p0 + pr;
        const int b  = p & (BB - 1);
        const int ij = p / BB;
        const int j  = ij & (NN - 1);
        const int i  = ij >> 7;
        const float entry = (mask[b * NN + i] && mask[b * NN + j]) ? 1.0f : 0.0f;
        const float val = (s + bout) * entry;
        if (mlp == 0) {
          atomicAdd(&esum[b], val);
        } else {
          const float* vh = vec_hat + (size_t)((i * NN + j) * BB + b) * 3;
          const float sc = val * INV_AVG;
          float* dst = out + 2 + (size_t)(b * NN + j) * 3;
          atomicAdd(&dst[0], sc * vh[0]);
          atomicAdd(&dst[1], sc * vh[1]);
          atomicAdd(&dst[2], sc * vh[2]);
        }
      }
    }
    __syncthreads();
  }

  if (tid < BB) atomicAdd(&out[tid], esum[tid] * INV_AVG2);
}

// ---------------- host side ----------------

extern "C" void kernel_launch(void* const* d_in, const int* in_sizes, int n_in,
                              void* d_out, int out_size, void* d_ws, size_t ws_size,
                              hipStream_t stream) {
  const float* x       = (const float*)d_in[0];
  const float* dist    = (const float*)d_in[1];
  const float* vec_hat = (const float*)d_in[2];
  const unsigned char* mask = (const unsigned char*)d_in[3];
  const float* rbf_W   = (const float*)d_in[4];
  const float* rbf_b   = (const float*)d_in[5];
  const float* e_Win   = (const float*)d_in[6];
  const float* e_bin   = (const float*)d_in[7];
  const float* e_Wh    = (const float*)d_in[8];
  const float* e_bh    = (const float*)d_in[9];
  const float* e_Wout  = (const float*)d_in[10];
  const float* e_bout  = (const float*)d_in[11];
  const float* f_Win   = (const float*)d_in[12];
  const float* f_bin   = (const float*)d_in[13];
  const float* f_Wh    = (const float*)d_in[14];
  const float* f_bh    = (const float*)d_in[15];
  const float* f_Wout  = (const float*)d_in[16];
  const float* f_bout  = (const float*)d_in[17];

  // workspace layout (bf16 transposed weights)
  char* ws = (char*)d_ws;
  size_t off = 0;
  bf16* eWinT = (bf16*)(ws + off); off += (size_t)HH * KF * 2;
  bf16* eWhT  = (bf16*)(ws + off); off += (size_t)LL * HH * HH * 2;
  bf16* eWo   = (bf16*)(ws + off); off += (size_t)HH * 2;
  bf16* fWinT = (bf16*)(ws + off); off += (size_t)HH * KF * 2;
  bf16* fWhT  = (bf16*)(ws + off); off += (size_t)LL * HH * HH * 2;
  bf16* fWo   = (bf16*)(ws + off); off += (size_t)HH * 2;
  (void)ws_size; (void)n_in; (void)in_sizes;

  const int thr = 256;
  transpose_to_bf16<<<(KF * HH + thr - 1) / thr, thr, 0, stream>>>(e_Win, eWinT, KF, HH);
  transpose_to_bf16<<<(KF * HH + thr - 1) / thr, thr, 0, stream>>>(f_Win, fWinT, KF, HH);
  for (int l = 0; l < LL; ++l) {
    transpose_to_bf16<<<(HH * HH + thr - 1) / thr, thr, 0, stream>>>(
        e_Wh + (size_t)l * HH * HH, eWhT + (size_t)l * HH * HH, HH, HH);
    transpose_to_bf16<<<(HH * HH + thr - 1) / thr, thr, 0, stream>>>(
        f_Wh + (size_t)l * HH * HH, fWhT + (size_t)l * HH * HH, HH, HH);
  }
  convert_to_bf16<<<(HH + thr - 1) / thr, thr, 0, stream>>>(e_Wout, eWo, HH);
  convert_to_bf16<<<(HH + thr - 1) / thr, thr, 0, stream>>>(f_Wout, fWo, HH);

  zero_f32<<<(out_size + thr - 1) / thr, thr, 0, stream>>>((float*)d_out, out_size);

  const size_t smem = (size_t)MT * KF * 2 + (size_t)MT * HH * 2 + 2 * sizeof(float);
  pair_mlp_kernel<<<NPAIR / MT, 256, smem, stream>>>(
      x, dist, vec_hat, mask, rbf_W, rbf_b,
      eWinT, e_bin, eWhT, e_bh, eWo, e_bout,
      fWinT, f_bin, fWhT, f_bh, fWo, f_bout,
      (float*)d_out);
}
